// GlobalSelfAttention_14620068675733
// MI455X (gfx1250) — matrix operand order repaired and checked by reading, not compiled, not run on hardware
//
#include <hip/hip_runtime.h>
#include <hip/hip_bf16.h>

// Problem constants
static constexpr int Bc = 4, Nc = 2048, Dc = 1024, Hc = 16, HDc = 64;

typedef __attribute__((ext_vector_type(16))) _Float16 v16h;
typedef __attribute__((ext_vector_type(8)))  float    v8f;

union V16H { v16h v; unsigned u[8]; };
union U4H8 { uint4 v; _Float16 h[8]; };

// ---------------------------------------------------------------------------
// f32 -> f16 conversion
// ---------------------------------------------------------------------------
__global__ __launch_bounds__(256) void cvt_f32_f16(const float* __restrict__ src,
                                                   _Float16* __restrict__ dst, int n) {
    int i = blockIdx.x * 256 + threadIdx.x;
    if (i < n) dst[i] = (_Float16)src[i];
}

// ---------------------------------------------------------------------------
// Shared WMMA GEMM core: C(64x128 block) = A(f16, row-major, lda=K) * B(f16, row-major, ldb=N)
// 256 threads = 8 waves, wave (waveM, waveN) in 2x4 grid, each wave does 32x32 via 2x2 WMMA.
// ---------------------------------------------------------------------------
__device__ __forceinline__ void gemm_core(const _Float16* __restrict__ A,
                                          const _Float16* __restrict__ Bm,
                                          int K, int ldb,
                                          int blockM, int blockN,
                                          _Float16* sA,   // [64][32]
                                          _Float16* sBT,  // [128][32]  (transposed B tile)
                                          v8f acc[2][2]) {
    const int tid   = threadIdx.x;
    const int lane  = tid & 31;
    const int wave  = tid >> 5;
    const int waveM = wave >> 2;   // 0..1
    const int waveN = wave & 3;    // 0..3
    const int lh    = lane >> 4;   // half-wave: 0 or 1
    const int l16   = lane & 15;

    for (int k0 = 0; k0 < K; k0 += 32) {
        __syncthreads();
        // Stage A tile 64x32 (each thread: one 16B load/store)
        {
            int m  = tid >> 2;
            int k8 = (tid & 3) << 3;
            uint4 t = *(const uint4*)(A + (size_t)(blockM + m) * K + k0 + k8);
            *(uint4*)(sA + m * 32 + k8) = t;
        }
        // Stage B tile 32x128 transposed into sBT[128][32]
        #pragma unroll
        for (int j = 0; j < 2; ++j) {
            int s  = tid + j * 256;       // 0..511
            int kk = s >> 4;              // 0..31
            int n8 = (s & 15) << 3;       // 0..120
            U4H8 t; t.v = *(const uint4*)(Bm + (size_t)(k0 + kk) * ldb + blockN + n8);
            #pragma unroll
            for (int e = 0; e < 8; ++e) sBT[(n8 + e) * 32 + kk] = t.h[e];
        }
        __syncthreads();

        V16H af[2], bf[2];
        #pragma unroll
        for (int ti = 0; ti < 2; ++ti) {
            int m = waveM * 32 + ti * 16 + l16;
            #pragma unroll
            for (int i = 0; i < 8; ++i) {
                int kp = (i < 4 ? 2 * i : 16 + 2 * (i - 4)) + 8 * lh;
                af[ti].u[i] = *(const unsigned*)(sA + m * 32 + kp);
            }
        }
        #pragma unroll
        for (int tj = 0; tj < 2; ++tj) {
            int n = waveN * 32 + tj * 16 + l16;
            #pragma unroll
            for (int i = 0; i < 8; ++i) {
                int kp = (i < 4 ? 2 * i : 16 + 2 * (i - 4)) + 8 * lh;
                bf[tj].u[i] = *(const unsigned*)(sBT + n * 32 + kp);
            }
        }
        #pragma unroll
        for (int ti = 0; ti < 2; ++ti)
            #pragma unroll
            for (int tj = 0; tj < 2; ++tj)
                acc[ti][tj] = __builtin_amdgcn_wmma_f32_16x16x32_f16(
                    false, af[ti].v, false, bf[tj].v, (short)0, acc[ti][tj], false, false);
    }
}

// ---------------------------------------------------------------------------
// Kernel 1: qkv = x @ Wqkv + bqkv, scattered into q/k/v [B,H,N,HD] f16 (q scaled)
// ---------------------------------------------------------------------------
__global__ __launch_bounds__(256) void qkv_gemm_kernel(const _Float16* __restrict__ xh,
                                                       const _Float16* __restrict__ wh,
                                                       const float* __restrict__ bias,
                                                       _Float16* __restrict__ qh,
                                                       _Float16* __restrict__ kh,
                                                       _Float16* __restrict__ vh) {
    __shared__ _Float16 sA[64 * 32];
    __shared__ _Float16 sBT[128 * 32];
    v8f acc[2][2];
    #pragma unroll
    for (int a = 0; a < 2; ++a)
        #pragma unroll
        for (int b = 0; b < 2; ++b)
            #pragma unroll
            for (int e = 0; e < 8; ++e) acc[a][b][e] = 0.0f;

    const int blockM = blockIdx.y * 64;
    const int blockN = blockIdx.x * 128;
    gemm_core(xh, wh, Dc, 3 * Dc, blockM, blockN, sA, sBT, acc);

    const int lane = threadIdx.x & 31;
    const int wave = threadIdx.x >> 5;
    const int waveM = wave >> 2, waveN = wave & 3;
    const int lh = lane >> 4, l16 = lane & 15;

    #pragma unroll
    for (int ti = 0; ti < 2; ++ti)
        #pragma unroll
        for (int tj = 0; tj < 2; ++tj)
            #pragma unroll
            for (int i = 0; i < 8; ++i) {
                int row = blockM + waveM * 32 + ti * 16 + i + 8 * lh;
                int col = blockN + waveN * 32 + tj * 16 + l16;
                float val = acc[ti][tj][i] + bias[col];
                int which = col >> 10;        // 0=q 1=k 2=v
                int d  = col & 1023;
                int hh = d >> 6;
                int hd = d & 63;
                int bb = row >> 11;
                int nn = row & 2047;
                size_t idx = (((size_t)(bb * Hc + hh)) * Nc + nn) * HDc + hd;
                if (which == 0)      qh[idx] = (_Float16)(val * 0.125f); // 1/sqrt(64)
                else if (which == 1) kh[idx] = (_Float16)val;
                else                 vh[idx] = (_Float16)val;
            }
}

// ---------------------------------------------------------------------------
// Kernel 2: flash attention. Block = 128 threads (4 waves), each wave: 16 queries.
// Grid: (N/64, H, B). K/V staged per 32-key chunk; online softmax; P@V via WMMA.
// ---------------------------------------------------------------------------
__global__ __launch_bounds__(128) void attn_kernel(const _Float16* __restrict__ qg,
                                                   const _Float16* __restrict__ kg,
                                                   const _Float16* __restrict__ vg,
                                                   _Float16* __restrict__ ag) {
    __shared__ _Float16 sK[32 * 64];        // [key][hd]
    __shared__ _Float16 sVT[64 * 32];       // [hd][key]
    __shared__ _Float16 sP[4][16 * 32];     // per-wave P tile [qrow][key]

    const int tid  = threadIdx.x;
    const int lane = tid & 31;
    const int wave = tid >> 5;          // 0..3
    const int lh   = lane >> 4;
    const int l16  = lane & 15;

    const int bb = blockIdx.z;
    const int hh = blockIdx.y;
    const int qBase = blockIdx.x * 64 + wave * 16;
    const size_t headOff = ((size_t)(bb * Hc + hh)) * Nc * HDc;
    const _Float16* qp = qg + headOff;
    const _Float16* kp = kg + headOff;
    const _Float16* vp = vg + headOff;

    // Load Q fragments (A layout), 2 k-chunks over HD=64
    V16H qa[2];
    {
        int q = qBase + l16;
        #pragma unroll
        for (int c = 0; c < 2; ++c)
            #pragma unroll
            for (int i = 0; i < 8; ++i) {
                int hd = (i < 4 ? 2 * i : 16 + 2 * (i - 4)) + 8 * lh + 32 * c;
                qa[c].u[i] = *(const unsigned*)(qp + (size_t)q * HDc + hd);
            }
    }

    float rmax[8], rsum[8];
    v8f o[4];
    #pragma unroll
    for (int i = 0; i < 8; ++i) { rmax[i] = -3.0e38f; rsum[i] = 0.0f; }
    #pragma unroll
    for (int t = 0; t < 4; ++t)
        #pragma unroll
        for (int e = 0; e < 8; ++e) o[t][e] = 0.0f;

    for (int keyBase = 0; keyBase < Nc; keyBase += 32) {
        __syncthreads();
        // Cooperative stage: K row-major, V transposed
        #pragma unroll
        for (int j = 0; j < 2; ++j) {
            int s   = tid + j * 128;     // 0..255
            int key = s >> 3;            // 0..31
            int hd8 = (s & 7) << 3;      // 0..56
            uint4 kv = *(const uint4*)(kp + (size_t)(keyBase + key) * HDc + hd8);
            *(uint4*)(sK + key * 64 + hd8) = kv;
            U4H8 vv; vv.v = *(const uint4*)(vp + (size_t)(keyBase + key) * HDc + hd8);
            #pragma unroll
            for (int e = 0; e < 8; ++e) sVT[(hd8 + e) * 32 + key] = vv.h[e];
        }
        __syncthreads();

        // S = Q @ K^T (16 x 32), two 16-key n-tiles, two 32-deep hd chunks each
        v8f sacc[2];
        #pragma unroll
        for (int nt = 0; nt < 2; ++nt) {
            #pragma unroll
            for (int e = 0; e < 8; ++e) sacc[nt][e] = 0.0f;
            #pragma unroll
            for (int c = 0; c < 2; ++c) {
                V16H kb;
                #pragma unroll
                for (int i = 0; i < 8; ++i)
                    kb.u[i] = *(const unsigned*)(sK + (nt * 16 + l16) * 64 +
                                                 (i < 4 ? 2 * i : 16 + 2 * (i - 4)) + 8 * lh + 32 * c);
                sacc[nt] = __builtin_amdgcn_wmma_f32_16x16x32_f16(
                    false, qa[c].v, false, kb.v, (short)0, sacc[nt], false, false);
            }
        }

        // Online softmax: row i lives across the 16 lanes of each half-wave
        #pragma unroll
        for (int i = 0; i < 8; ++i) {
            float vmax = fmaxf(sacc[0][i], sacc[1][i]);
            #pragma unroll
            for (int off = 1; off < 16; off <<= 1)
                vmax = fmaxf(vmax, __shfl_xor(vmax, off, 32));
            float nm = fmaxf(rmax[i], vmax);
            float sc = __expf(rmax[i] - nm);
            rmax[i] = nm;
            float p0 = __expf(sacc[0][i] - nm);
            float p1 = __expf(sacc[1][i] - nm);
            float ps = p0 + p1;
            #pragma unroll
            for (int off = 1; off < 16; off <<= 1)
                ps += __shfl_xor(ps, off, 32);
            rsum[i] = rsum[i] * sc + ps;
            #pragma unroll
            for (int t = 0; t < 4; ++t) o[t][i] *= sc;
            // store P (C-layout element) to LDS to re-layout into A fragments
            sP[wave][(i + 8 * lh) * 32 + l16]      = (_Float16)p0;
            sP[wave][(i + 8 * lh) * 32 + 16 + l16] = (_Float16)p1;
        }

        // P(16x32) @ V(32x64): 4 n-tiles of 16 hd columns
        V16H pa;
        #pragma unroll
        for (int i = 0; i < 8; ++i) {
            int kp2 = (i < 4 ? 2 * i : 16 + 2 * (i - 4)) + 8 * lh;
            pa.u[i] = *(const unsigned*)(&sP[wave][l16 * 32 + kp2]);
        }
        #pragma unroll
        for (int t = 0; t < 4; ++t) {
            V16H vb;
            #pragma unroll
            for (int i = 0; i < 8; ++i)
                vb.u[i] = *(const unsigned*)(sVT + (t * 16 + l16) * 32 + (i < 4 ? 2 * i : 16 + 2 * (i - 4)) + 8 * lh);
            o[t] = __builtin_amdgcn_wmma_f32_16x16x32_f16(
                false, pa.v, false, vb.v, (short)0, o[t], false, false);
        }
    }

    // Normalize and write attention output as [B, N, D] f16 (D index = h*64+hd)
    #pragma unroll
    for (int i = 0; i < 8; ++i) rsum[i] = 1.0f / rsum[i];
    #pragma unroll
    for (int t = 0; t < 4; ++t)
        #pragma unroll
        for (int i = 0; i < 8; ++i) {
            int q  = qBase + i + 8 * lh;
            int hd = t * 16 + l16;
            ag[((size_t)bb * Nc + q) * Dc + hh * HDc + hd] = (_Float16)(o[t][i] * rsum[i]);
        }
}

// ---------------------------------------------------------------------------
// Kernel 3: out = attn_out @ Wout + bout  (f32 output)
// ---------------------------------------------------------------------------
__global__ __launch_bounds__(256) void out_gemm_kernel(const _Float16* __restrict__ ah,
                                                       const _Float16* __restrict__ wh,
                                                       const float* __restrict__ bias,
                                                       float* __restrict__ out) {
    __shared__ _Float16 sA[64 * 32];
    __shared__ _Float16 sBT[128 * 32];
    v8f acc[2][2];
    #pragma unroll
    for (int a = 0; a < 2; ++a)
        #pragma unroll
        for (int b = 0; b < 2; ++b)
            #pragma unroll
            for (int e = 0; e < 8; ++e) acc[a][b][e] = 0.0f;

    const int blockM = blockIdx.y * 64;
    const int blockN = blockIdx.x * 128;
    gemm_core(ah, wh, Dc, Dc, blockM, blockN, sA, sBT, acc);

    const int lane = threadIdx.x & 31;
    const int wave = threadIdx.x >> 5;
    const int waveM = wave >> 2, waveN = wave & 3;
    const int lh = lane >> 4, l16 = lane & 15;

    #pragma unroll
    for (int ti = 0; ti < 2; ++ti)
        #pragma unroll
        for (int tj = 0; tj < 2; ++tj)
            #pragma unroll
            for (int i = 0; i < 8; ++i) {
                int row = blockM + waveM * 32 + ti * 16 + i + 8 * lh;
                int col = blockN + waveN * 32 + tj * 16 + l16;
                out[(size_t)row * Dc + col] = acc[ti][tj][i] + bias[col];
            }
}

// ---------------------------------------------------------------------------
// Host launcher
// ---------------------------------------------------------------------------
extern "C" void kernel_launch(void* const* d_in, const int* in_sizes, int n_in,
                              void* d_out, int out_size, void* d_ws, size_t ws_size,
                              hipStream_t stream) {
    const float* x    = (const float*)d_in[0];
    const float* Wqkv = (const float*)d_in[1];
    const float* bqkv = (const float*)d_in[2];
    const float* Wout = (const float*)d_in[3];
    const float* bout = (const float*)d_in[4];
    float* out = (float*)d_out;

    char* ws = (char*)d_ws;
    size_t off = 0;
    auto wsalloc = [&](size_t bytes) -> void* {
        void* p = ws + off;
        off += (bytes + 255) & ~(size_t)255;
        return p;
    };
    const size_t nX  = (size_t)Bc * Nc * Dc;          // 8.4M
    const size_t nWq = (size_t)Dc * 3 * Dc;           // 3.1M
    const size_t nWo = (size_t)Dc * Dc;               // 1.0M
    const size_t nQ  = (size_t)Bc * Hc * Nc * HDc;    // 8.4M

    _Float16* xh  = (_Float16*)wsalloc(nX * 2);
    _Float16* wqh = (_Float16*)wsalloc(nWq * 2);
    _Float16* woh = (_Float16*)wsalloc(nWo * 2);
    _Float16* qh  = (_Float16*)wsalloc(nQ * 2);
    _Float16* kh  = (_Float16*)wsalloc(nQ * 2);
    _Float16* vh  = (_Float16*)wsalloc(nQ * 2);
    _Float16* ah  = (_Float16*)wsalloc(nX * 2);

    // 1) convert inputs to f16
    cvt_f32_f16<<<(int)((nX  + 255) / 256), 256, 0, stream>>>(x,    xh,  (int)nX);
    cvt_f32_f16<<<(int)((nWq + 255) / 256), 256, 0, stream>>>(Wqkv, wqh, (int)nWq);
    cvt_f32_f16<<<(int)((nWo + 255) / 256), 256, 0, stream>>>(Wout, woh, (int)nWo);

    // 2) QKV projection: M=8192, N=3072 -> grid (24, 128)
    qkv_gemm_kernel<<<dim3(3 * Dc / 128, Bc * Nc / 64), 256, 0, stream>>>(
        xh, wqh, bqkv, qh, kh, vh);

    // 3) flash attention: grid (N/64, H, B), 128 threads
    attn_kernel<<<dim3(Nc / 64, Hc, Bc), 128, 0, stream>>>(qh, kh, vh, ah);

    // 4) output projection: M=8192, N=1024 -> grid (8, 128)
    out_gemm_kernel<<<dim3(Dc / 128, Bc * Nc / 64), 256, 0, stream>>>(
        ah, woh, bout, out);
}
